// TransferFunctionSegmentGenerator_35347580846654
// MI455X (gfx1250) — matrix operand order repaired and checked by reading, not compiled, not run on hardware
//
#include <hip/hip_runtime.h>
#include <math.h>
#include <stdint.h>

// ---------------------------------------------------------------------------
// CDNA5 / gfx1250 implementation of TransferFunctionSegmentGenerator
// ---------------------------------------------------------------------------
typedef __attribute__((ext_vector_type(16))) _Float16 v16h;
typedef __attribute__((ext_vector_type(2)))  _Float16 v2h;
typedef __attribute__((ext_vector_type(8)))  float    v8f;

#define MODEL_DIM 128
#define N_FRAMES  128
#define WINDOW    512
#define HOP       256
#define N_SAMPLES 32768
#define N_COEFFS  257
#define BATCH     256
#define NBT       32768          // BATCH * N_FRAMES  (bt dimension)
#define KIMAG     260            // imag block starts here in padded-K layout
#define KPAD      544            // 520 (260 real + 260 imag) padded to mult of 32
#define NKSTEP    (KPAD / 32)    // 17
#define M2        16384          // half-length complex FFT size
#define SPEC_STRIDE 16448        // float2 per batch row (>= 16385)
#define WTF_N     65792          // N_COEFFS*2*N_FRAMES

// ---- workspace layout (bytes) ----
#define OFF_XH     ((size_t)0)                       // 256*128 f16        =   64 KB
#define OFF_ENV128 ((size_t)65536)                   // 256*128 f32        =  128 KB
#define OFF_ENVSIG ((size_t)196608)                  // 256*32768 f32      =   32 MB
#define OFF_AFFT   ((size_t)33751040)                // 544*32768 f16      = 34.0 MB
#define OFF_BASIS  ((size_t)69402624)                // 512*544 f16        =  544 KB
#define OFF_WIN    ((size_t)69959680)                // 512 f32
#define OFF_SIG    ((size_t)69961728)                // 256*32768 f32      =   32 MB
#define OFF_SPEC   ((size_t)103516160)               // 256*16448 float2   = 33.7 MB

// ===========================================================================
// init: Hamming window, irfft ortho basis (cos / -sin), zero padded Afft rows
// ===========================================================================
__global__ void k_init(float* __restrict__ win,
                       _Float16* __restrict__ basis,
                       _Float16* __restrict__ afft) {
  int blk = blockIdx.x;
  if (blk < WINDOW) {
    int n = blk;
    if (threadIdx.x == 0)
      win[n] = 0.54f - 0.46f * cosf(6.283185307179586f * (float)n / 512.0f);
    const float inv = 0.044194173824159216f;  // 1/sqrt(512)
    for (int k = (int)threadIdx.x; k < KPAD; k += 256) {
      float v = 0.0f;
      if (k < 2 * KIMAG) {
        int c = (k < KIMAG) ? k : (k - KIMAG);
        if (c < N_COEFFS) {
          float ang = 6.283185307179586f * (float)c * (float)n / 512.0f;
          if (k < KIMAG) {
            float alpha = (c == 0 || c == 256) ? 1.0f : 2.0f;
            v = alpha * cosf(ang) * inv;
          } else {
            v = -2.0f * sinf(ang) * inv;   // auto-zero at c=0,256
          }
        }
      }
      basis[(size_t)n * KPAD + k] = (_Float16)v;
    }
  } else {
    int z = blk - WINDOW;                    // 0..29 -> rows 257..259, 517..543
    int row = (z < 3) ? (257 + z) : (514 + z);
    for (int j = (int)threadIdx.x; j < NBT; j += 256)
      afft[(size_t)row * NBT + j] = (_Float16)0.0f;
  }
}

// x (f32) -> f16 for WMMA A-operand
__global__ void k_xh(const float* __restrict__ x, _Float16* __restrict__ xh) {
  int i = blockIdx.x * 256 + threadIdx.x;
  xh[i] = (_Float16)x[i];
}

// env128[b,t] = (x@W_env + b_env)^2   (tiny GEMM, plain f32 dot)
__global__ void k_env128(const float* __restrict__ x,
                         const float* __restrict__ Wenv,
                         const float* __restrict__ benv,
                         float* __restrict__ env128) {
  int b = blockIdx.x, t = threadIdx.x;
  float acc = benv[t];
  for (int d = 0; d < MODEL_DIM; ++d)
    acc = fmaf(x[b * MODEL_DIM + d], Wenv[d * N_FRAMES + t], acc);
  env128[b * N_FRAMES + t] = acc * acc;
}

// envsig[b,s] = lerp(env128[b], s) * noise[s]
__global__ void k_envsig(const float* __restrict__ env128,
                         const float* __restrict__ noise,
                         float* __restrict__ envsig) {
  int idx = blockIdx.x * 256 + threadIdx.x;
  int b = idx >> 15;
  int s = idx & (N_SAMPLES - 1);
  float pos = ((float)s + 0.5f) * (1.0f / 256.0f) - 0.5f;
  pos = fminf(fmaxf(pos, 0.0f), 127.0f);
  int lo = (int)floorf(pos);
  int hi = min(lo + 1, 127);
  float w = pos - (float)lo;
  float e = env128[b * 128 + lo] * (1.0f - w) + env128[b * 128 + hi] * w;
  envsig[idx] = e * noise[s];
}

// ===========================================================================
// tf GEMM (WMMA) + fused pair-normalization.
// Block: 256 thr (8 waves). Grid: (257 coeffs, 8 t-tiles).
// Each wave: 2 batch tiles of 16; accumulates the real and imag 16x16 tiles
// for coefficient c (columns c*256+t and c*256+128+t), K=128 in 4 steps.
// ===========================================================================
__global__ __launch_bounds__(256) void
k_tf(const _Float16* __restrict__ xh, const float* __restrict__ Wtf,
     const float* __restrict__ btf, _Float16* __restrict__ afft) {
  __shared__ _Float16 ldsw[2][4][16][32];   // [real/imag][kstep][n][k%32]
  int c  = blockIdx.x;            // 0..256
  int tt = blockIdx.y;            // 0..7
  int tid = threadIdx.x;
  int jbase = c * 256 + tt * 16;

  // stage W_tf tile (f32 -> f16), both real and imag column groups
  for (int it = 0; it < 16; ++it) {
    int flat = it * 256 + tid;           // 0..4095
    int n  = flat & 15;
    int d  = (flat >> 4) & 127;
    int ri = flat >> 11;
    float v = Wtf[(size_t)d * WTF_N + jbase + ri * 128 + n];
    ldsw[ri][d >> 5][n][d & 31] = (_Float16)v;
  }
  __syncthreads();

  int wave = tid >> 5, lane = tid & 31;
  int h = lane >> 4, mn = lane & 15;
  float br = btf[c * 256 + tt * 16 + mn];
  float bi = btf[c * 256 + 128 + tt * 16 + mn];

  for (int rep = 0; rep < 2; ++rep) {
    int btile = wave * 2 + rep;          // 0..15
    v8f accr = {}; v8f acci = {};
    for (int ks = 0; ks < 4; ++ks) {
      // A fragment: 16-bit A 16x32, half h gets contiguous K {16h..16h+15}
      v16h a;
      int arow = (btile * 16 + mn) * MODEL_DIM + ks * 32;
#pragma unroll
      for (int v = 0; v < 8; ++v) {
        int k0 = 16 * h + 2 * v;
        v2h p = *(const v2h*)(xh + arow + k0);
        a[2 * v] = p[0]; a[2 * v + 1] = p[1];
      }
      // B fragments (lane = N, per-VGPR K-pair {2v+16h, 2v+1+16h})
      v16h wbr, wbi;
#pragma unroll
      for (int v = 0; v < 8; ++v) {
        int kk = 16 * h + 2 * v;
        v2h pr = *(const v2h*)(&ldsw[0][ks][mn][kk]);
        v2h pi = *(const v2h*)(&ldsw[1][ks][mn][kk]);
        wbr[2 * v] = pr[0]; wbr[2 * v + 1] = pr[1];
        wbi[2 * v] = pi[0]; wbi[2 * v + 1] = pi[1];
      }
      accr = __builtin_amdgcn_wmma_f32_16x16x32_f16(false, a, false, wbr,
                                                    (short)0, accr, false, false);
      acci = __builtin_amdgcn_wmma_f32_16x16x32_f16(false, a, false, wbi,
                                                    (short)0, acci, false, false);
    }
    // fused normalization: tf / (norm+1e-8) * clip(norm, 0, 0.9999)
    int t = tt * 16 + mn;
#pragma unroll
    for (int r = 0; r < 8; ++r) {
      int b = btile * 16 + r + 8 * h;    // C/D layout: M = r + 8*(lane>=16)
      float nr = accr[r] + br;
      float ni = acci[r] + bi;
      float nrm = sqrtf(nr * nr + ni * ni);
      float sc = fminf(nrm, 0.9999f) / (nrm + 1e-8f);
      size_t bt = (size_t)b * N_FRAMES + t;
      afft[(size_t)c * NBT + bt]           = (_Float16)(nr * sc);
      afft[(size_t)(KIMAG + c) * NBT + bt] = (_Float16)(ni * sc);
    }
  }
}

// ===========================================================================
// irfft-as-GEMM (WMMA) + fused Hamming window + overlap-add.
// frames(512 x 32768bt) = Basis(512 x 544) x Afft(544 x 32768).
// Block: 256 thr (8 waves), grid 2048 bt-tiles; each wave owns 4 M-tiles.
// B tiles are DMA'd global->LDS with GLOBAL_LOAD_ASYNC_TO_LDS_B32 and
// double-buffered so the async engine overlaps the WMMA pipes (ASYNCcnt).
// ===========================================================================
__global__ __launch_bounds__(256) void
k_irfft(const _Float16* __restrict__ basis, const _Float16* __restrict__ afft,
        const float* __restrict__ win, float* __restrict__ sig) {
  __shared__ _Float16 ldsb[2][32][16];    // [buf][k][j] : 2 x 1 KB, DMA target
  int btt = blockIdx.x;                   // 0..2047
  int jb = btt * 16;
  int tid = threadIdx.x;
  int wave = tid >> 5, lane = tid & 31;
  int h = lane >> 4, mn = lane & 15;
  v8f acc[4] = {{}, {}, {}, {}};

  // per-thread async-copy slice: one b32 (4 bytes = 2 halfs) of the 1 KB tile
  unsigned lds_base = (unsigned)(size_t)(&ldsb[0][0][0]) + (unsigned)(tid * 4);
  int krow = tid >> 3;                    // 0..31
  int jcol = (tid & 7) << 1;              // 0,2,..,14

  // prologue: DMA K-step 0 into buffer 0
  {
    const _Float16* g = afft + (size_t)krow * NBT + jb + jcol;
    asm volatile("global_load_async_to_lds_b32 %0, %1, off"
                 :: "v"(lds_base), "v"((uint64_t)(size_t)g) : "memory");
  }
  asm volatile("s_wait_asynccnt 0x0" ::: "memory");
  __syncthreads();

  for (int ks = 0; ks < NKSTEP; ++ks) {
    int cur = ks & 1;
    if (ks + 1 < NKSTEP) {                 // kick DMA for next K-step (overlaps WMMA)
      const _Float16* g = afft + (size_t)((ks + 1) * 32 + krow) * NBT + jb + jcol;
      unsigned dst = lds_base + (unsigned)((cur ^ 1) * 1024);
      asm volatile("global_load_async_to_lds_b32 %0, %1, off"
                   :: "v"(dst), "v"((uint64_t)(size_t)g) : "memory");
    }

    v16h bfrag;                            // per-lane K-pairs from [k][j] tile
#pragma unroll
    for (int v = 0; v < 8; ++v) {
      int kk = 16 * h + 2 * v;
      bfrag[2 * v]     = ldsb[cur][kk][mn];
      bfrag[2 * v + 1] = ldsb[cur][kk + 1][mn];
    }
#pragma unroll
    for (int mt = 0; mt < 4; ++mt) {
      int mtile = wave + 8 * mt;           // 0..31
      v16h afrag;
      int arow = (mtile * 16 + mn) * KPAD + ks * 32;
#pragma unroll
      for (int v = 0; v < 8; ++v) {
        int k0 = 16 * h + 2 * v;
        v2h p = *(const v2h*)(basis + arow + k0);
        afrag[2 * v] = p[0]; afrag[2 * v + 1] = p[1];
      }
      acc[mt] = __builtin_amdgcn_wmma_f32_16x16x32_f16(false, afrag, false, bfrag,
                                                       (short)0, acc[mt], false, false);
    }

    asm volatile("s_wait_asynccnt 0x0" ::: "memory");
    __syncthreads();
  }

  // window + overlap-add: sample s = t*HOP + n  (two contributors per s)
  int bt = jb + mn, b = bt >> 7, t = bt & 127;
#pragma unroll
  for (int mt = 0; mt < 4; ++mt) {
    int mtile = wave + 8 * mt;
#pragma unroll
    for (int r = 0; r < 8; ++r) {
      int ns = mtile * 16 + r + 8 * h;     // 0..511
      int s = t * HOP + ns;
      if (s < N_SAMPLES)
        atomicAdd(&sig[(size_t)b * N_SAMPLES + s], acc[mt][r] * win[ns]);
    }
  }
}

// ===========================================================================
// 16384-point complex radix-2 DIT FFT fully in LDS (128 KB of 320 KB WGP LDS)
// ===========================================================================
__device__ __forceinline__ void fft16384(float* re, float* im, int tid) {
  for (int s = 1; s <= 14; ++s) {
    int half = 1 << (s - 1);
    float base = -3.14159265358979323846f / (float)half;   // forward: e^{-i*}
    for (int t = tid; t < M2 / 2; t += 1024) {
      int j  = t & (half - 1);
      int i0 = ((t >> (s - 1)) << s) + j;
      int i1 = i0 + half;
      float ws, wc;
      __sincosf(base * (float)j, &ws, &wc);
      float xr = re[i1], xi = im[i1];
      float tr = wc * xr - ws * xi;
      float ti = wc * xi + ws * xr;
      float ur = re[i0], ui = im[i0];
      re[i1] = ur - tr; im[i1] = ui - ti;
      re[i0] = ur + tr; im[i0] = ui + ti;
    }
    __syncthreads();
  }
}

// forward rfft via even/odd packing; mode 0: spec=X, mode 1: spec *= X
__global__ __launch_bounds__(1024) void
k_fft_fwd(const float* __restrict__ src, float2* __restrict__ spec, int mode) {
  __shared__ float re[M2];
  __shared__ float im[M2];
  int b = blockIdx.x, tid = threadIdx.x;
  const float* p = src + (size_t)b * N_SAMPLES;
  for (int i = tid; i < M2; i += 1024) {
    int r = __brev((unsigned)i) >> 18;         // 14-bit bit-reverse
    re[r] = p[2 * i]; im[r] = p[2 * i + 1];
  }
  __syncthreads();
  fft16384(re, im, tid);
  float2* sp = spec + (size_t)b * SPEC_STRIDE;
  for (int k = tid; k <= M2; k += 1024) {
    int k1 = k & (M2 - 1);
    int k2 = (M2 - k) & (M2 - 1);
    float zr1 = re[k1], zi1 = im[k1], zr2 = re[k2], zi2 = im[k2];
    float er = 0.5f * (zr1 + zr2), ei = 0.5f * (zi1 - zi2);
    float orr = 0.5f * (zi1 + zi2), oi = -0.5f * (zr1 - zr2);
    float sw, cw;
    __sincosf(-6.283185307179586f * (float)k / (float)N_SAMPLES, &sw, &cw);
    float Xr = er + cw * orr - sw * oi;
    float Xi = ei + cw * oi + sw * orr;
    if (mode) {
      float2 E = sp[k];
      sp[k] = make_float2(Xr * E.x - Xi * E.y, Xr * E.y + Xi * E.x);
    } else {
      sp[k] = make_float2(Xr, Xi);
    }
  }
}

// inverse C2R of Hermitian product, scaled by 1/sqrt(N) (ortho conv factor)
__global__ __launch_bounds__(1024) void
k_fft_inv(const float2* __restrict__ spec, float* __restrict__ out) {
  __shared__ float re[M2];
  __shared__ float im[M2];
  int b = blockIdx.x, tid = threadIdx.x;
  const float2* sp = spec + (size_t)b * SPEC_STRIDE;
  for (int k = tid; k < M2; k += 1024) {
    float2 Pk = sp[k];
    float2 Pm = sp[M2 - k];
    float er = 0.5f * (Pk.x + Pm.x), ei = 0.5f * (Pk.y - Pm.y);
    float dr = 0.5f * (Pk.x - Pm.x), di = 0.5f * (Pk.y + Pm.y);
    float sw, cw;
    __sincosf(6.283185307179586f * (float)k / (float)N_SAMPLES, &sw, &cw); // W^{-k}
    float or_ = dr * cw - di * sw;
    float oi  = dr * sw + di * cw;
    // Z = Xe + i*Xo ; store conj(Z) bit-reversed (inverse via conjugate trick)
    int r = __brev((unsigned)k) >> 18;
    re[r] = er - oi;
    im[r] = -(ei + or_);
  }
  __syncthreads();
  fft16384(re, im, tid);
  const float scale = 1.0f / (16384.0f * 181.01933598375618f);  // 1/(M*sqrt(N))
  float* o = out + (size_t)b * N_SAMPLES;
  for (int n = tid; n < M2; n += 1024) {
    o[2 * n]     =  re[n] * scale;
    o[2 * n + 1] = -im[n] * scale;
  }
}

// ===========================================================================
extern "C" void kernel_launch(void* const* d_in, const int* in_sizes, int n_in,
                              void* d_out, int out_size, void* d_ws, size_t ws_size,
                              hipStream_t stream) {
  (void)in_sizes; (void)n_in; (void)out_size; (void)ws_size;
  const float* x    = (const float*)d_in[0];
  const float* Wenv = (const float*)d_in[1];
  const float* benv = (const float*)d_in[2];
  const float* Wtf  = (const float*)d_in[3];
  const float* btf  = (const float*)d_in[4];
  const float* nois = (const float*)d_in[5];
  float* out = (float*)d_out;

  char* w = (char*)d_ws;
  _Float16* xh     = (_Float16*)(w + OFF_XH);
  float*    env128 = (float*)   (w + OFF_ENV128);
  float*    envsig = (float*)   (w + OFF_ENVSIG);
  _Float16* afft   = (_Float16*)(w + OFF_AFFT);
  _Float16* basis  = (_Float16*)(w + OFF_BASIS);
  float*    win    = (float*)   (w + OFF_WIN);
  float*    sig    = (float*)   (w + OFF_SIG);
  float2*   spec   = (float2*)  (w + OFF_SPEC);

  hipMemsetAsync(sig, 0, (size_t)BATCH * N_SAMPLES * sizeof(float), stream);

  k_init  <<<dim3(WINDOW + 30), dim3(256), 0, stream>>>(win, basis, afft);
  k_xh    <<<dim3(128),  dim3(256), 0, stream>>>(x, xh);
  k_env128<<<dim3(BATCH), dim3(128), 0, stream>>>(x, Wenv, benv, env128);
  k_envsig<<<dim3(BATCH * N_SAMPLES / 256), dim3(256), 0, stream>>>(env128, nois, envsig);

  k_tf    <<<dim3(N_COEFFS, 8), dim3(256), 0, stream>>>(xh, Wtf, btf, afft);
  k_irfft <<<dim3(NBT / 16),    dim3(256), 0, stream>>>(basis, afft, win, sig);

  k_fft_fwd<<<dim3(BATCH), dim3(1024), 0, stream>>>(envsig, spec, 0);
  k_fft_fwd<<<dim3(BATCH), dim3(1024), 0, stream>>>(sig,    spec, 1);
  k_fft_inv<<<dim3(BATCH), dim3(1024), 0, stream>>>(spec, out);
}
